// InteractingLayer_87162066305063
// MI455X (gfx1250) — matrix operand order repaired and checked by reading, not compiled, not run on hardware
//
#include <hip/hip_runtime.h>

// ---------------------------------------------------------------------------
// InteractingLayer (multi-head field attention) for MI455X / gfx1250.
// B=4096, F=64, E=256, H=4, DH=64.  One workgroup (8 wave32) per batch elem.
// All six GEMMs via v_wmma_f32_16x16x32_bf16.
// ---------------------------------------------------------------------------

typedef __attribute__((ext_vector_type(16))) __bf16 bf16x16;
typedef __attribute__((ext_vector_type(8)))  __bf16 bf16x8;
typedef __attribute__((ext_vector_type(8)))  float  v8f;
typedef unsigned short u16;
typedef unsigned int   u32;

#define B_SZ 4096
#define F_SZ 64
#define E_SZ 256
#define H_SZ 4

__device__ __forceinline__ u16 f32_bf16(float f) {
  union { float f; u32 u; } v; v.f = f;
  u32 r = v.u + 0x7FFFu + ((v.u >> 16) & 1u);   // round-to-nearest-even
  return (u16)(r >> 16);
}

// A fragment (16x32 bf16): lane: M=lane&15, K = half*16 + i  (contiguous 32B)
__device__ __forceinline__ bf16x16 ld_a(const u16* base, int stride, int m0, int k0) {
  const int lane = threadIdx.x & 31;
  const int half = lane >> 4;
  return *(const bf16x16*)(base + (m0 + (lane & 15)) * stride + k0 + half * 16);
}

// B fragment (32x16 bf16): lane: N=lane&15, K = half*16 + i  (contiguous 32B)
__device__ __forceinline__ bf16x16 ld_b(const u16* base, int stride, int n0, int k0) {
  const int lane = threadIdx.x & 31;
  const int half = lane >> 4;
  return *(const bf16x16*)(base + (n0 + (lane & 15)) * stride + k0 + half * 16);
}

__device__ __forceinline__ v8f wmma_bf16(bf16x16 a, bf16x16 b, v8f c) {
  return __builtin_amdgcn_wmma_f32_16x16x32_bf16(false, a, false, b, (short)0, c,
                                                 false, false);
}

// C/D tile (16x16 f32): VGPR r holds M = r + 8*half, N = lane&15.
__device__ __forceinline__ void st_cd_rm(u16* base, int stride, int m0, int n0, v8f c) {
  const int lane = threadIdx.x & 31;
  u16* p = base + (m0 + (lane >> 4) * 8) * stride + n0 + (lane & 15);
#pragma unroll
  for (int r = 0; r < 8; ++r) p[(size_t)r * stride] = f32_bf16(c[r]);
}

// Transposed store: dest[n][m]; each lane writes 8 contiguous bf16 (b128).
__device__ __forceinline__ void st_cd_tr(u16* base, int stride, int m0, int n0, v8f c) {
  const int lane = threadIdx.x & 31;
  u16* p = base + (n0 + (lane & 15)) * stride + m0 + (lane >> 4) * 8;
  u32 w0 = (u32)f32_bf16(c[0]) | ((u32)f32_bf16(c[1]) << 16);
  u32 w1 = (u32)f32_bf16(c[2]) | ((u32)f32_bf16(c[3]) << 16);
  u32 w2 = (u32)f32_bf16(c[4]) | ((u32)f32_bf16(c[5]) << 16);
  u32 w3 = (u32)f32_bf16(c[6]) | ((u32)f32_bf16(c[7]) << 16);
  *(uint4*)p = make_uint4(w0, w1, w2, w3);
}

// ---------------------------------------------------------------------------
// Kernel 0: fp32 weights -> bf16, transposed to [n][k] for contiguous B frags.
// ---------------------------------------------------------------------------
__global__ void convert_weights_kernel(const float* __restrict__ wq,
                                       const float* __restrict__ wk,
                                       const float* __restrict__ wv,
                                       const float* __restrict__ wr,
                                       u16* __restrict__ out) {
  int idx = blockIdx.x * blockDim.x + threadIdx.x;   // 4 * 256 * 256 total
  int m = idx >> 16;
  int r = idx & 65535;
  int k = r >> 8;
  int n = r & 255;
  const float* src = (m == 0) ? wq : (m == 1) ? wk : (m == 2) ? wv : wr;
  out[(size_t)m * 65536 + (size_t)n * 256 + k] = f32_bf16(src[(size_t)k * 256 + n]);
}

// ---------------------------------------------------------------------------
// Kernel 1: fused projections + attention + softmax + residual + ReLU.
// Dynamic LDS layout (ushort units unless noted):
//   sh_xp [64][256]  : x_b bf16 (stage1)  ->  P probs [4][64][64] bf16 (stage2/3)
//   sh_q  [64][256]  : Q bf16 row-major
//   sh_k  [64][256]  : K bf16 row-major
//   sh_vt [256][64]  : V bf16 transposed (Vt[e][f])
//   sh_s  [64][64]f32: per-head score scratch
// Total 144 KB.
// ---------------------------------------------------------------------------
__global__ void __launch_bounds__(256)
interacting_layer_kernel(const float* __restrict__ x,
                         const u16* __restrict__ wt,
                         float* __restrict__ out) {
  extern __shared__ unsigned char smem_raw[];
  u16* sh_xp = (u16*)smem_raw;
  u16* sh_q  = sh_xp + 16384;
  u16* sh_k  = sh_q  + 16384;
  u16* sh_vt = sh_k  + 16384;
  float* sh_s = (float*)(sh_vt + 16384);

  const int b    = blockIdx.x;
  const int tid  = threadIdx.x;
  const int wv   = tid >> 5;
  const int lane = tid & 31;

  // ---- load x_b (64x256 f32) -> LDS bf16 ----
  const float* xb = x + (size_t)b * (F_SZ * E_SZ);
#pragma unroll
  for (int i = 0; i < 16; ++i) {
    int j = i * 256 + tid;                      // float4 index, 4096 total
    float4 v = ((const float4*)xb)[j];
    u32 lo = (u32)f32_bf16(v.x) | ((u32)f32_bf16(v.y) << 16);
    u32 hi = (u32)f32_bf16(v.z) | ((u32)f32_bf16(v.w) << 16);
    ((uint2*)sh_xp)[j] = make_uint2(lo, hi);
  }
  __syncthreads();

  const u16* wtq = wt;
  const u16* wtk = wt + 65536;
  const u16* wtv = wt + 2 * 65536;
  const u16* wtr = wt + 3 * 65536;

  // ---- stage 1: Q,K,V,R projections (E=256 -> 8 WMMA k-steps per tile) ----
  // Wave w owns output tiles p = w*8 .. w*8+7  (p = mt*16 + nt, mt:F/16, nt:E/16)
  v8f rAcc[8];
#pragma unroll
  for (int pl = 0; pl < 8; ++pl) {
    const int p  = wv * 8 + pl;
    const int mt = p >> 4;
    const int nt = p & 15;
    v8f aq = {}, ak = {}, av = {}, ar = {};
#pragma unroll
    for (int kt = 0; kt < 8; ++kt) {
      bf16x16 a = ld_a(sh_xp, 256, mt * 16, kt * 32);
      aq = wmma_bf16(a, ld_b(wtq, 256, nt * 16, kt * 32), aq);
      ak = wmma_bf16(a, ld_b(wtk, 256, nt * 16, kt * 32), ak);
      av = wmma_bf16(a, ld_b(wtv, 256, nt * 16, kt * 32), av);
      ar = wmma_bf16(a, ld_b(wtr, 256, nt * 16, kt * 32), ar);
    }
    st_cd_rm(sh_q, 256, mt * 16, nt * 16, aq);
    st_cd_rm(sh_k, 256, mt * 16, nt * 16, ak);
    st_cd_tr(sh_vt, 64, mt * 16, nt * 16, av);   // Vt[e][f]
    rAcc[pl] = ar;                               // residual proj kept in regs
  }
  __syncthreads();

  // ---- stage 2: per head: S = Q Kt (64x64), softmax rows, P -> LDS bf16 ----
#pragma unroll 1
  for (int h = 0; h < H_SZ; ++h) {
#pragma unroll
    for (int tt = 0; tt < 2; ++tt) {           // 16 tiles, 2 per wave
      const int t  = wv * 2 + tt;
      const int mi = t >> 2;
      const int mj = t & 3;
      v8f acc = {};
#pragma unroll
      for (int kt = 0; kt < 2; ++kt) {         // K-dim = DH = 64
        bf16x16 a  = ld_a(sh_q, 256, mi * 16, h * 64 + kt * 32);
        bf16x16 bb = ld_b(sh_k, 256, mj * 16, h * 64 + kt * 32);
        acc = wmma_bf16(a, bb, acc);
      }
      const int n  = mj * 16 + (lane & 15);
      const int m0 = mi * 16 + (lane >> 4) * 8;
#pragma unroll
      for (int r = 0; r < 8; ++r) sh_s[(m0 + r) * 64 + n] = acc[r];
    }
    __syncthreads();

    if (tid < 64) {                            // waves 0-1 exactly (uniform)
      float mx = -3.402823466e38f;
      for (int j = 0; j < 64; ++j) mx = fmaxf(mx, sh_s[tid * 64 + j]);
      float sum = 0.f;
      for (int j = 0; j < 64; ++j) {
        float e = __expf(sh_s[tid * 64 + j] - mx);
        sh_s[tid * 64 + j] = e;
        sum += e;
      }
      const float inv = 1.f / sum;
      u16* prow = sh_xp + h * 4096 + tid * 64;   // P overlays dead x region
      for (int j = 0; j < 64; j += 2) {
        u32 pk = (u32)f32_bf16(sh_s[tid * 64 + j] * inv) |
                 ((u32)f32_bf16(sh_s[tid * 64 + j + 1] * inv) << 16);
        *(u32*)(prow + j) = pk;
      }
    }
    __syncthreads();
  }

  // ---- stage 3: out = relu(P @ V + R), WMMA seeded with R accumulator ----
  float* ob = out + (size_t)b * (F_SZ * E_SZ);
#pragma unroll
  for (int pl = 0; pl < 8; ++pl) {
    const int p  = wv * 8 + pl;
    const int mt = p >> 4;
    const int nt = p & 15;
    const int h  = nt >> 2;
    v8f acc = rAcc[pl];
#pragma unroll
    for (int kt = 0; kt < 2; ++kt) {           // K-dim = F = 64
      bf16x16 a  = ld_a(sh_xp + h * 4096, 64, mt * 16, kt * 32);  // P_h
      bf16x16 bb = ld_b(sh_vt, 64, nt * 16, kt * 32);             // Vt
      acc = wmma_bf16(a, bb, acc);
    }
    const int n  = nt * 16 + (lane & 15);
    const int m0 = mt * 16 + (lane >> 4) * 8;
#pragma unroll
    for (int r = 0; r < 8; ++r)
      ob[(size_t)(m0 + r) * 256 + n] = fmaxf(acc[r], 0.f);
  }
}

// ---------------------------------------------------------------------------
extern "C" void kernel_launch(void* const* d_in, const int* in_sizes, int n_in,
                              void* d_out, int out_size, void* d_ws, size_t ws_size,
                              hipStream_t stream) {
  (void)in_sizes; (void)n_in; (void)out_size; (void)ws_size;
  const float* x  = (const float*)d_in[0];
  const float* wq = (const float*)d_in[1];
  const float* wk = (const float*)d_in[2];
  const float* wv = (const float*)d_in[3];
  const float* wr = (const float*)d_in[4];
  u16* wt = (u16*)d_ws;                        // 4 * 256*256 bf16 = 512 KB

  convert_weights_kernel<<<1024, 256, 0, stream>>>(wq, wk, wv, wr, wt);

  const size_t shmem = 144 * 1024;
  hipFuncSetAttribute(reinterpret_cast<const void*>(interacting_layer_kernel),
                      hipFuncAttributeMaxDynamicSharedMemorySize, (int)shmem);
  interacting_layer_kernel<<<dim3(B_SZ), dim3(256), shmem, stream>>>(
      x, wt, (float*)d_out);
}
